// PropagationGraphConvLayer_nBatch_78950088835521
// MI455X (gfx1250) — matrix operand order repaired and checked by reading, not compiled, not run on hardware
//
#include <hip/hip_runtime.h>
#include <hip/hip_bf16.h>

// ---------------------------------------------------------------------------
// PropagationGraphConv forward on MI455X (gfx1250, wave32, WMMA + TDM).
// GEMMs: bf16 WMMA (v_wmma_f32_16x16x32_bf16), f32 accumulate.
// Tile staging: Tensor Data Mover (tensor_load_to_lds), double-buffered so
// the DMA engine overlaps the WMMA stream; TENSORcnt + one barrier per step.
// Graph scatter: f32 global atomics (keeps the irregular reduction exact).
// ---------------------------------------------------------------------------

typedef __attribute__((ext_vector_type(16))) __bf16 v16bf;
typedef __attribute__((ext_vector_type(8)))  __bf16 v8bf;
typedef __attribute__((ext_vector_type(8)))  float  v8f;

#define NB      16
#define NNODES  4096
#define NEDGES  4096
#define MROWS   (NB * NNODES)      /* 65536 */
#define BE      (NB * NEDGES)      /* 65536 */
#define DF      512                /* FIN == HID == OUT */

// ------------------------- elementwise helpers -----------------------------

__global__ void convert_f32_bf16_kernel(const float* __restrict__ src,
                                        __bf16* __restrict__ dst, int n4) {
    int i = blockIdx.x * blockDim.x + threadIdx.x;
    if (i >= n4) return;
    const float4 v = reinterpret_cast<const float4*>(src)[i];
    __bf16* p = dst + (long)i * 4;
    p[0] = (__bf16)v.x; p[1] = (__bf16)v.y; p[2] = (__bf16)v.z; p[3] = (__bf16)v.w;
}

// WT[n*K + k] = bf16(W[k*N + n])  -- so WMMA B fragments are contiguous reads
__global__ void transpose_w_kernel(const float* __restrict__ W,
                                   __bf16* __restrict__ WT, int K, int N) {
    int idx = blockIdx.x * blockDim.x + threadIdx.x;
    if (idx >= K * N) return;
    int k = idx / N, n = idx - k * N;
    WT[(long)n * K + k] = (__bf16)W[idx];
}

__global__ void fill_bias_kernel(float* __restrict__ h,
                                 const float* __restrict__ b, int total) {
    int idx = blockIdx.x * blockDim.x + threadIdx.x;
    if (idx >= total) return;
    h[idx] = b[idx & (DF - 1)];
}

// h[row,:] += w * support[col,:]   (one block per edge; f32 atomics)
__global__ void scatter_edges_kernel(const float* __restrict__ support,
                                     const float* __restrict__ vals,
                                     const int* __restrict__ rows,
                                     const int* __restrict__ cols,
                                     float* __restrict__ h) {
    const int e   = blockIdx.x;
    const int row = rows[e];
    const int col = cols[e];
    const float w = vals[e];
    const float* s = support + (long)col * DF;
    float*       d = h       + (long)row * DF;
    for (int j = threadIdx.x; j < DF; j += blockDim.x)
        atomicAdd(&d[j], w * s[j]);
}

// z[i] = relu([h1[i] | X[root(i)]])  (bf16, stride 1024)
__global__ void build_z_kernel(const float* __restrict__ h1,
                               const float* __restrict__ feat,
                               const int* __restrict__ root,
                               __bf16* __restrict__ z) {
    int idx = blockIdx.x * blockDim.x + threadIdx.x;
    if (idx >= MROWS * DF) return;
    int i = idx >> 9, j = idx & (DF - 1);
    int r = root[i >> 12];                       // global root row of graph
    float a = h1[idx];
    float b = feat[(long)r * DF + j];
    z[(long)i * (2 * DF) + j]      = (__bf16)(a > 0.f ? a : 0.f);
    z[(long)i * (2 * DF) + DF + j] = (__bf16)(b > 0.f ? b : 0.f);
}

// zz[i] = [relu(h2pre[i]) | h1[root(i)]]   (rh is NOT relu'd in the reference)
__global__ void build_zz_kernel(const float* __restrict__ h2pre,
                                const float* __restrict__ h1,
                                const int* __restrict__ root,
                                __bf16* __restrict__ zz) {
    int idx = blockIdx.x * blockDim.x + threadIdx.x;
    if (idx >= MROWS * DF) return;
    int i = idx >> 9, j = idx & (DF - 1);
    int r = root[i >> 12];
    float a = h2pre[idx];
    zz[(long)i * (2 * DF) + j]      = (__bf16)(a > 0.f ? a : 0.f);
    zz[(long)i * (2 * DF) + DF + j] = (__bf16)h1[(long)r * DF + j];
}

// ----------------------------- WMMA GEMM -----------------------------------
// C[M x 512] = A[M x K](bf16,row-major) * B (given as BT[512 x K] bf16)
// Block: 256 threads = 8 waves; block tile 128(M) x 128(N); K step = 32.
// Wave (wid&3 -> M strip of 32, wid>>2 -> N strip of 64): 2x4 f32 accs,
// so each B fragment feeds two WMMAs.  mode 1: out = leaky_relu(acc+bias).

__device__ __forceinline__ v16bf ld_frag(const __bf16* p, int gap) {
    v8bf lo = *(const v8bf*)p;
    v8bf hi = *(const v8bf*)(p + gap);
    v16bf r;
#pragma unroll
    for (int i = 0; i < 8; ++i) { r[i] = lo[i]; r[i + 8] = hi[i]; }
    return r;
}

#define LDSA 40   /* padded row stride (halfwords): 32 data + 8 pad */

// one K-step of math: 2 A frags x 4 B frags -> 8 WMMAs
__device__ __forceinline__ void wmma_step(const __bf16* Asb, const __bf16* Bsb,
                                          int wm, int wn, int half, int l15,
                                          v8f (&acc)[2][4]) {
    // A frag: elems 0..7 -> k = 8*half + i ; elems 8..15 -> +16 (ISA layout)
    v16bf a0 = ld_frag(&Asb[(wm + l15) * LDSA + half * 8], 16);
    v16bf a1 = ld_frag(&Asb[(wm + 16 + l15) * LDSA + half * 8], 16);
#pragma unroll
    for (int j = 0; j < 4; ++j) {
        // B frag: elems 0..7 -> k = 8*half + i ; elems 8..15 -> +16, lane = n column
        v16bf b = ld_frag(&Bsb[(wn + j * 16 + l15) * LDSA + half * 8], 16);
        acc[0][j] = __builtin_amdgcn_wmma_f32_16x16x32_bf16(
            false, a0, false, b, (short)0, acc[0][j], false, false);
        acc[1][j] = __builtin_amdgcn_wmma_f32_16x16x32_bf16(
            false, a1, false, b, (short)0, acc[1][j], false, false);
    }
}

#if __has_builtin(__builtin_amdgcn_tensor_load_to_lds) && \
    __has_builtin(__builtin_amdgcn_s_wait_tensorcnt)
#define USE_TDM 1
typedef __attribute__((ext_vector_type(4))) unsigned int u32x4;
typedef __attribute__((ext_vector_type(4))) int          i32x4;
typedef __attribute__((ext_vector_type(8))) int          i32x8;

// DMA one 128-row x 32-elem bf16 tile (row stride = K elems) into LDS,
// inserting 8 halfwords of padding after every row (16dw data + 4dw pad)
// to reproduce the LDSA=40 layout in hardware.
__device__ __forceinline__ void tdm_load_tile(const __bf16* gsrc, unsigned lds_addr,
                                              int K, int tensor_rows) {
    unsigned long long ga = (unsigned long long)(size_t)gsrc;
    u32x4 g0;
    g0[0] = 1u;                                        // count=1, user D#
    g0[1] = lds_addr;                                  // LDS byte address
    g0[2] = (unsigned)(ga & 0xFFFFFFFFu);              // global_addr lo
    g0[3] = (unsigned)((ga >> 32) & 0x01FFFFFFu) | (2u << 30); // addr hi | type=2
    i32x8 g1;
    g1[0] = (1 << 16)   /* data_size = 2B */
          | (1 << 20)   /* pad_enable  */
          | (3 << 22)   /* pad_interval = 16 dwords */
          | (3 << 25);  /* pad_amount   = 4 dwords  */
    g1[1] = (K & 0xFFFF) << 16;                        // tensor_dim0 lo16
    g1[2] = ((unsigned)K >> 16) | ((tensor_rows & 0xFFFF) << 16); // dim0 hi | dim1 lo
    g1[3] = ((unsigned)tensor_rows >> 16) | (32 << 16);// dim1 hi | tile_dim0=32
    g1[4] = 128;                                       // tile_dim1=128, tile_dim2=0
    g1[5] = K;                                         // tensor_dim0_stride lo
    g1[6] = 0;
    g1[7] = 0;
    i32x4 z4 = {0, 0, 0, 0};
#if defined(__clang_major__) && (__clang_major__ >= 23)
    i32x8 z8 = {0, 0, 0, 0, 0, 0, 0, 0};
    __builtin_amdgcn_tensor_load_to_lds(g0, g1, z4, z4, z8, 0);
#else
    __builtin_amdgcn_tensor_load_to_lds(g0, g1, z4, z4, 0);
#endif
}
#endif

__global__ void __launch_bounds__(256)
gemm_bf16_wmma_kernel(const __bf16* __restrict__ A,
                      const __bf16* __restrict__ BT,
                      float* __restrict__ C, int K, int Mtot,
                      const float* __restrict__ bias, int mode) {
    __shared__ __bf16 As[2][128 * LDSA];   // double-buffered tiles
    __shared__ __bf16 Bs[2][128 * LDSA];

    const int tid   = threadIdx.x;
    const int lane  = tid & 31;
    const int wid   = tid >> 5;
    const int half  = lane >> 4;        // lane half within wave32
    const int l15   = lane & 15;
    const int wm    = (wid & 3) * 32;   // wave M offset in block tile
    const int wn    = (wid >> 2) * 64;  // wave N offset in block tile
    const long blkM = (long)blockIdx.x * 128;
    const long blkN = (long)blockIdx.y * 128;

    v8f acc[2][4];
#pragma unroll
    for (int mi = 0; mi < 2; ++mi)
#pragma unroll
        for (int j = 0; j < 4; ++j) acc[mi][j] = v8f{};

#ifdef USE_TDM
    const unsigned lds_a0 = (unsigned)(size_t)&As[0][0]; // flat->LDS: low 32 bits
    const unsigned lds_a1 = (unsigned)(size_t)&As[1][0];
    const unsigned lds_b0 = (unsigned)(size_t)&Bs[0][0];
    const unsigned lds_b1 = (unsigned)(size_t)&Bs[1][0];

    // prologue: DMA tiles for k0 = 0 into buffer 0
    if (wid == 0) {
        tdm_load_tile(&A[blkM * K],  lds_a0, K, Mtot);
        tdm_load_tile(&BT[blkN * K], lds_b0, K, DF);
    }
    __builtin_amdgcn_s_wait_tensorcnt(0);
    __syncthreads();

    int cur = 0;
    for (int k0 = 0; k0 < K; k0 += 32, cur ^= 1) {
        // kick next tiles into the other buffer; DMA overlaps the WMMA chain
        if (k0 + 32 < K && wid == 0) {
            tdm_load_tile(&A[blkM * K + k0 + 32],  cur ? lds_a0 : lds_a1, K, Mtot);
            tdm_load_tile(&BT[blkN * K + k0 + 32], cur ? lds_b0 : lds_b1, K, DF);
        }
        wmma_step(&As[cur][0], &Bs[cur][0], wm, wn, half, l15, acc);
        __builtin_amdgcn_s_wait_tensorcnt(0);  // next buffer landed (wave 0)
        __syncthreads();                       // publish next / retire current
    }
#else
    for (int k0 = 0; k0 < K; k0 += 32) {
        __syncthreads();                 // previous compute done, LDS reusable
#pragma unroll
        for (int it = 0; it < 4; ++it) { // 1024 x 16B chunks over 256 threads
            int idx = tid + it * 256;
            int row = (idx >> 2) & 127, ch = idx & 3;
            if (idx < 512)
                *(v8bf*)&As[0][row * LDSA + ch * 8] =
                    *(const v8bf*)&A[(blkM + row) * K + k0 + ch * 8];
            else
                *(v8bf*)&Bs[0][row * LDSA + ch * 8] =
                    *(const v8bf*)&BT[(blkN + row) * K + k0 + ch * 8];
        }
        if (k0 + 32 < K) {
            __builtin_prefetch(&A[(blkM + (tid >> 1)) * K + k0 + 32], 0, 1);
            __builtin_prefetch(&BT[(blkN + (tid >> 1)) * K + k0 + 32], 0, 1);
        }
        __syncthreads();                 // LDS tiles visible to all waves
        wmma_step(&As[0][0], &Bs[0][0], wm, wn, half, l15, acc);
    }
#endif

    // C layout: element r -> m = r + 8*half, n = lane&15
#pragma unroll
    for (int mi = 0; mi < 2; ++mi) {
        const long gm0 = blkM + wm + mi * 16 + half * 8;
#pragma unroll
        for (int j = 0; j < 4; ++j) {
            const long gn = blkN + wn + j * 16 + l15;
#pragma unroll
            for (int r = 0; r < 8; ++r) {
                float v = acc[mi][j][r];
                if (mode == 1) {
                    v += bias[gn];
                    v = v > 0.f ? v : 0.01f * v;   // leaky_relu(0.01)
                }
                C[(gm0 + r) * DF + gn] = v;
            }
        }
    }
}

// ------------------------------- driver ------------------------------------

extern "C" void kernel_launch(void* const* d_in, const int* in_sizes, int n_in,
                              void* d_out, int out_size, void* d_ws, size_t ws_size,
                              hipStream_t stream) {
    const float* feat = (const float*)d_in[0];
    const int*   adjs = (const int*)d_in[1];    // [2, B*E] global indices
    const float* vals = (const float*)d_in[2];
    const int*   root = (const int*)d_in[3];    // global root row per graph
    const float* W1   = (const float*)d_in[6];
    const float* b1   = (const float*)d_in[7];
    const float* W2   = (const float*)d_in[8];
    const float* b2   = (const float*)d_in[9];
    const float* Wl   = (const float*)d_in[10];
    const float* bl   = (const float*)d_in[11];

    char* ws = (char*)d_ws;
    // 128 MB bf16 staging (reused: Xbf -> z -> zz), 3 x 128 MB f32, weights
    __bf16* stage   = (__bf16*)(ws);
    float*  support = (float*)(ws + 0x08000000);
    float*  h1      = (float*)(ws + 0x10000000);
    float*  h2      = (float*)(ws + 0x18000000);
    __bf16* WT1     = (__bf16*)(ws + 0x20000000);
    __bf16* WT2     = (__bf16*)(ws + 0x20000000 + (1 << 19));
    __bf16* WTl     = (__bf16*)(ws + 0x20000000 + (1 << 19) + (1 << 20));
    float*  out     = (float*)d_out;

    const int elems = MROWS * DF;                      // 33.5M
    const dim3 blk(256);
    const dim3 gemm_grid(MROWS / 128, DF / 128);       // 512 x 4

    // X -> bf16
    convert_f32_bf16_kernel<<<(elems / 4 + 255) / 256, blk, 0, stream>>>(feat, stage, elems / 4);
    // W -> bf16 transposed
    transpose_w_kernel<<<(DF * DF + 255) / 256, blk, 0, stream>>>(W1, WT1, DF, DF);
    transpose_w_kernel<<<(2 * DF * DF + 255) / 256, blk, 0, stream>>>(W2, WT2, 2 * DF, DF);
    transpose_w_kernel<<<(2 * DF * DF + 255) / 256, blk, 0, stream>>>(Wl, WTl, 2 * DF, DF);

    // conv1: support1 = Xbf @ W1
    gemm_bf16_wmma_kernel<<<gemm_grid, blk, 0, stream>>>(stage, WT1, support, DF, MROWS, nullptr, 0);
    fill_bias_kernel<<<(elems + 255) / 256, blk, 0, stream>>>(h1, b1, elems);
    scatter_edges_kernel<<<BE, blk, 0, stream>>>(support, vals, adjs, adjs + BE, h1);

    // z = relu([h1 | x_root]) -> bf16
    build_z_kernel<<<(elems + 255) / 256, blk, 0, stream>>>(h1, feat, root, stage);

    // conv2: support2 = z @ W2
    gemm_bf16_wmma_kernel<<<gemm_grid, blk, 0, stream>>>(stage, WT2, support, 2 * DF, MROWS, nullptr, 0);
    fill_bias_kernel<<<(elems + 255) / 256, blk, 0, stream>>>(h2, b2, elems);
    scatter_edges_kernel<<<BE, blk, 0, stream>>>(support, vals, adjs, adjs + BE, h2);

    // zz = [relu(h2) | h1_root] -> bf16
    build_zz_kernel<<<(elems + 255) / 256, blk, 0, stream>>>(h2, h1, root, stage);

    // linear: out = leaky_relu(zz @ Wl + bl)
    gemm_bf16_wmma_kernel<<<gemm_grid, blk, 0, stream>>>(stage, WTl, out, 2 * DF, MROWS, bl, 1);
}
